// GNNAttentionBlock_43499428774564
// MI455X (gfx1250) — matrix operand order repaired and checked by reading, not compiled, not run on hardware
//
#include <hip/hip_runtime.h>
#include <hip/hip_bf16.h>

typedef __attribute__((ext_vector_type(16))) _Float16 v16h;
typedef __attribute__((ext_vector_type(8)))  _Float16 v8h;
typedef __attribute__((ext_vector_type(8)))  float    v8f;

#define B_GRAPHS 64
#define N_NODES  512
#define DIM      128
#define HEADS    4
#define DHEAD    32
#define E_EDGES  8192
#define TOTAL_NODES (B_GRAPHS * N_NODES)          // 32768
#define TOTAL_EDGES (B_GRAPHS * E_EDGES)          // 524288
#define INNER    (HEADS * DHEAD)                  // 128

__device__ __forceinline__ v16h pack16(v8h a, v8h b) {
  v16h r;
#pragma unroll
  for (int i = 0; i < 8; ++i) { r[i] = a[i]; r[i + 8] = b[i]; }
  return r;
}

__device__ __forceinline__ v8f wmma_f16(v16h a, v16h b, v8f c) {
  // D = A(16x32 f16) x B(32x16 f16) + C(16x16 f32)
  return __builtin_amdgcn_wmma_f32_16x16x32_f16(
      /*neg_a=*/false, a, /*neg_b=*/false, b,
      /*c_mod=*/(short)0, c, /*reuse_a=*/false, /*reuse_b=*/false);
}

// Wave-relative LDS byte offset = low 32 bits of the flat address (ISA 10.2).
__device__ __forceinline__ unsigned lds_off(const void* p) {
  return (unsigned)(unsigned long long)p;
}

// Async DMA: 16B global -> LDS per lane, tracked by ASYNCcnt (ISA 15.18.3 op 98).
__device__ __forceinline__ void async_load_b128(unsigned lds_byte_off, const void* g) {
  asm volatile("global_load_async_to_lds_b128 %0, %1, off"
               :: "v"(lds_byte_off), "v"((unsigned long long)g)
               : "memory");
}

__device__ __forceinline__ void wait_async_le2() {
  asm volatile("s_wait_asynccnt 2" ::: "memory");
}
__device__ __forceinline__ void wait_async_0() {
  asm volatile("s_wait_asynccnt 0" ::: "memory");
}

// ---------------------------------------------------------------- prep kernels

__global__ void cvt_f32_to_f16_kernel(const float* __restrict__ in,
                                      _Float16* __restrict__ out, int n) {
  int i = blockIdx.x * 256 + threadIdx.x;
  if (i < n) out[i] = (_Float16)in[i];
}

// W: din x dout (row-major, fp32)  ->  Wt: dout x din (row-major, f16)
__global__ void prep_wt_kernel(const float* __restrict__ W,
                               _Float16* __restrict__ Wt, int din, int dout) {
  int i = blockIdx.x * 256 + threadIdx.x;
  if (i >= din * dout) return;
  int r = i / dout, c = i % dout;
  Wt[c * din + r] = (_Float16)W[i];
}

// ---------------------------------------------------------------- generic WMMA GEMM
// C(MxN, f32) = A(MxK, f16 row-major) * Wt(NxK, f16 row-major)^T + bias(N)
// One wave computes one 16x16 tile; blockDim = 128 (4 waves).
__global__ void gemm_f16_wmma_bias_kernel(const _Float16* __restrict__ A,
                                          const _Float16* __restrict__ Wt,
                                          const float* __restrict__ bias,
                                          float* __restrict__ C,
                                          int M, int N, int K) {
  const int lane = threadIdx.x & 31;
  const int wid  = threadIdx.x >> 5;
  const int hi   = lane >> 4;      // half-wave select
  const int ln   = lane & 15;
  const int tile = blockIdx.x * 4 + wid;
  const int ntn  = N >> 4;
  const int mtile = tile / ntn;
  const int ntile = tile % ntn;
  const int arow = mtile * 16 + ln;   // A-fragment: M = lane%16
  const int bcol = ntile * 16 + ln;   // B-fragment: N = lane%16

  v8f acc = {0.f, 0.f, 0.f, 0.f, 0.f, 0.f, 0.f, 0.f};
#pragma unroll
  for (int kc = 0; kc < 128; kc += 32) {
    // A fragment 16x32 f16: half e -> K = kc + e + 16*hi, contiguous per lane
    const _Float16* ap = A + (size_t)arow * K + kc + 16 * hi;
    v8h a0 = *(const v8h*)ap;
    v8h a1 = *(const v8h*)(ap + 8);
    // B fragment 32x16 f16: half e -> K = kc + e + 16*hi, contiguous per lane
    v16h b = *(const v16h*)(Wt + (size_t)bcol * K + kc + 16 * hi);
    acc = wmma_f16(pack16(a0, a1), b, acc);
  }
  const float bv = bias[bcol];
  float* outp = C + (size_t)(mtile * 16 + 8 * hi) * N + bcol;
#pragma unroll
  for (int r = 0; r < 8; ++r) outp[(size_t)r * N] = acc[r] + bv;
}

// ---------------------------------------------------------------- graph kernels

__global__ void deg_count_kernel(const int* __restrict__ row,
                                 float* __restrict__ degcnt) {
  int e = blockIdx.x * 256 + threadIdx.x;
  if (e < TOTAL_EDGES) atomicAdd(&degcnt[row[e]], 1.0f);
}

// one wave (32 lanes) per edge, 4 floats per lane
__global__ void edge_msg_kernel(const int* __restrict__ ei,
                                const float* __restrict__ ee,
                                const float* __restrict__ h,
                                const float* __restrict__ degcnt,
                                float* __restrict__ agg) {
  int t = blockIdx.x * 256 + threadIdx.x;
  int e = t >> 5;
  int lane = t & 31;
  if (e >= TOTAL_EDGES) return;
  int r = ei[e];
  int c = ei[TOTAL_EDGES + e];
  float norm = rsqrtf(degcnt[r] + 1.0f) * rsqrtf(degcnt[c] + 1.0f);
  int d = lane * 4;
  float4 hv = *(const float4*)(h + (size_t)r * DIM + d);
  float4 ev = *(const float4*)(ee + (size_t)e * DIM + d);
  float m0 = norm * fmaxf(hv.x + ev.x, 0.f);
  float m1 = norm * fmaxf(hv.y + ev.y, 0.f);
  float m2 = norm * fmaxf(hv.z + ev.z, 0.f);
  float m3 = norm * fmaxf(hv.w + ev.w, 0.f);
  float* ap = agg + (size_t)c * DIM + d;
  atomicAdd(ap + 0, m0);
  atomicAdd(ap + 1, m1);
  atomicAdd(ap + 2, m2);
  atomicAdd(ap + 3, m3);
}

__global__ void combine_kernel(const float* __restrict__ h,
                               const float* __restrict__ agg,
                               const float* __restrict__ degcnt,
                               const float* __restrict__ root,
                               _Float16* __restrict__ nf16) {
  int i = blockIdx.x * 256 + threadIdx.x;
  if (i >= TOTAL_NODES * DIM) return;
  int node = i >> 7;
  int d = i & 127;
  float deg = degcnt[node] + 1.0f;
  float v = agg[i] + fmaxf(h[i] + root[d], 0.f) / deg;
  nf16[i] = (_Float16)v;
}

// ---------------------------------------------------------------- QKV GEMM
// q,k: [b,h,n,dh] f16 ; vt: [b,h,dh,n] f16 (transposed for the P@V B-fragment)
__global__ void gemm_qkv_kernel(const _Float16* __restrict__ nf,
                                const _Float16* __restrict__ Wqkv_t, // 384 x 128
                                _Float16* __restrict__ q,
                                _Float16* __restrict__ k,
                                _Float16* __restrict__ vt) {
  const int K = DIM;
  const int lane = threadIdx.x & 31;
  const int wid  = threadIdx.x >> 5;
  const int hi   = lane >> 4;
  const int ln   = lane & 15;
  const int tile = blockIdx.x * 4 + wid;
  const int mtile = tile / 24;
  const int ntile = tile % 24;
  const int arow = mtile * 16 + ln;
  const int bcol = ntile * 16 + ln;

  v8f acc = {0.f, 0.f, 0.f, 0.f, 0.f, 0.f, 0.f, 0.f};
#pragma unroll
  for (int kc = 0; kc < 128; kc += 32) {
    const _Float16* ap = nf + (size_t)arow * K + kc + 16 * hi;
    v8h a0 = *(const v8h*)ap;
    v8h a1 = *(const v8h*)(ap + 8);
    v16h b = *(const v16h*)(Wqkv_t + (size_t)bcol * K + kc + 16 * hi);
    acc = wmma_f16(pack16(a0, a1), b, acc);
  }
  // route: columns [0,128)=q, [128,256)=k, [256,384)=v (16-wide tiles never straddle)
  const int seg = ntile >> 3;           // 0=q 1=k 2=v
  const int hh  = (ntile & 7) >> 1;     // head
  const int dh  = ((ntile & 1) << 4) + ln;
#pragma unroll
  for (int r = 0; r < 8; ++r) {
    int row = mtile * 16 + r + 8 * hi;
    int b_ = row >> 9;
    int n_ = row & 511;
    int bh = b_ * HEADS + hh;
    _Float16 val = (_Float16)acc[r];
    if (seg == 0)      q[((size_t)bh * N_NODES + n_) * DHEAD + dh] = val;
    else if (seg == 1) k[((size_t)bh * N_NODES + n_) * DHEAD + dh] = val;
    else               vt[((size_t)bh * DHEAD + dh) * N_NODES + n_] = val;
  }
}

// ---------------------------------------------------------------- flash attention
// grid: (B*H) * 8 blocks, 128 threads (4 waves). Each wave: 16 query rows.
// K/V chunks (32 keys) are staged into LDS once per block with async DMA
// (double-buffered, ASYNCcnt-tracked) and shared by all 4 waves.
__global__ void attn_kernel(const _Float16* __restrict__ q,
                            const _Float16* __restrict__ k,
                            const _Float16* __restrict__ vt,
                            _Float16* __restrict__ out) {
  __shared__ __align__(32) _Float16 ktile[2][32 * 32];  // [key][dh]
  __shared__ __align__(32) _Float16 vtile[2][32 * 32];  // [dh][key]
  __shared__ __align__(32) _Float16 pst[4][16 * 32];
  const float SCALE = 0.17677669529663687f;  // 1/sqrt(32)
  const int tid  = threadIdx.x;
  const int lane = tid & 31;
  const int w    = tid >> 5;
  const int hi   = lane >> 4;
  const int ln   = lane & 15;
  const int bh   = blockIdx.x >> 3;
  const int rb   = ((blockIdx.x & 7) << 2) + w;  // 16-row block id, 0..31
  const int qrow = rb * 16 + ln;

  const _Float16* qb = q  + (size_t)bh * N_NODES * DHEAD;
  const char*     kg = (const char*)(k  + (size_t)bh * N_NODES * DHEAD);
  const char*     vg = (const char*)(vt + (size_t)bh * DHEAD * N_NODES);

  // per-thread staging addresses: each thread DMAs one 16B piece per tile
  const unsigned koff0 = lds_off(&ktile[0][0]) + tid * 16;
  const unsigned koff1 = lds_off(&ktile[1][0]) + tid * 16;
  const unsigned voff0 = lds_off(&vtile[0][0]) + tid * 16;
  const unsigned voff1 = lds_off(&vtile[1][0]) + tid * 16;
  // K chunk: 32 keys x 64B, fully contiguous (2KB) at byte k0*64
  // V chunk: 32 dh-rows x 64B, row stride 1024B, columns start at byte k0*2
  const char* kgt = kg + tid * 16;
  const char* vgt = vg + (size_t)(tid >> 2) * 1024 + (tid & 3) * 16;

  // Q A-fragment (16 rows x DH=32), pre-scaled by 1/sqrt(DH)
  v16h aq;
  {
    const _Float16* p = qb + (size_t)qrow * DHEAD + 16 * hi;
    v8h x0 = *(const v8h*)p;
    v8h x1 = *(const v8h*)(p + 8);
    aq = pack16(x0, x1);
#pragma unroll
    for (int i = 0; i < 16; ++i) aq[i] = (_Float16)((float)aq[i] * SCALE);
  }

  float m[8], l[8];
  v8f o1 = {0.f, 0.f, 0.f, 0.f, 0.f, 0.f, 0.f, 0.f};
  v8f o2 = {0.f, 0.f, 0.f, 0.f, 0.f, 0.f, 0.f, 0.f};
#pragma unroll
  for (int r = 0; r < 8; ++r) { m[r] = -1e30f; l[r] = 0.f; }

  // prologue: DMA chunk 0 into buffer 0
  async_load_b128(koff0, kgt);
  async_load_b128(voff0, vgt + 0);

  for (int it = 0; it < 16; ++it) {
    const int k0 = it * 32;
    const _Float16* kt = (it & 1) ? &ktile[1][0] : &ktile[0][0];
    const _Float16* vtl = (it & 1) ? &vtile[1][0] : &vtile[0][0];
    if (it + 1 < 16) {
      // DMA next chunk into the other buffer (readers of it finished at the
      // end-of-iteration barrier of it-1)
      const int kn = (it + 1) * 32;
      async_load_b128((it & 1) ? koff0 : koff1, kgt + (size_t)kn * 64);
      async_load_b128((it & 1) ? voff0 : voff1, vgt + (size_t)kn * 2);
      wait_async_le2();   // in-order completion -> current chunk has landed
    } else {
      wait_async_0();
    }
    __syncthreads();

    // scores: B fragments from LDS K tile [key][dh]: K=dh (=e+16*hi), N=key
    v16h bk0 = *(const v16h*)(kt + ln * 32 + 16 * hi);
    v16h bk1 = *(const v16h*)(kt + (16 + ln) * 32 + 16 * hi);
    v8f zc = {0.f, 0.f, 0.f, 0.f, 0.f, 0.f, 0.f, 0.f};
    v8f s0 = wmma_f16(aq, bk0, zc);
    v8f s1 = wmma_f16(aq, bk1, zc);

#pragma unroll
    for (int r = 0; r < 8; ++r) {
      float x0 = s0[r];
      float x1 = s1[r];
      float mx = fmaxf(x0, x1);
      mx = fmaxf(mx, __shfl_xor(mx, 1, 32));
      mx = fmaxf(mx, __shfl_xor(mx, 2, 32));
      mx = fmaxf(mx, __shfl_xor(mx, 4, 32));
      mx = fmaxf(mx, __shfl_xor(mx, 8, 32));
      float mn = fmaxf(m[r], mx);
      float sc = __expf(m[r] - mn);
      float p0 = __expf(x0 - mn);
      float p1 = __expf(x1 - mn);
      float rs = p0 + p1;
      rs += __shfl_xor(rs, 1, 32);
      rs += __shfl_xor(rs, 2, 32);
      rs += __shfl_xor(rs, 4, 32);
      rs += __shfl_xor(rs, 8, 32);
      l[r] = l[r] * sc + rs;
      m[r] = mn;
      o1[r] *= sc;
      o2[r] *= sc;
      // stage P (C-layout element -> row-major 16x32 LDS tile)
      pst[w][(r + 8 * hi) * 32 + ln]      = (_Float16)p0;
      pst[w][(r + 8 * hi) * 32 + 16 + ln] = (_Float16)p1;
    }
    // LDS write -> read within the same wave; DS ops are in-order per wave,
    // the wait + memory clobber pins compiler ordering.
    asm volatile("s_wait_dscnt 0" ::: "memory");

    // reload P as an A-fragment (16x32, K = key index within chunk)
    v16h ap;
    {
      const _Float16* pp = &pst[w][ln * 32 + 16 * hi];
      v8h x0 = *(const v8h*)pp;
      v8h x1 = *(const v8h*)(pp + 8);
      ap = pack16(x0, x1);
    }
    // V B-fragments from LDS V tile [dh][key]: B[key][dhcol], K=key (=e+16*hi)
    v16h bv0 = *(const v16h*)(vtl + ln * 32 + 16 * hi);
    v16h bv1 = *(const v16h*)(vtl + (16 + ln) * 32 + 16 * hi);
    o1 = wmma_f16(ap, bv0, o1);
    o2 = wmma_f16(ap, bv1, o2);

    __syncthreads();   // all reads done before this buffer is overwritten
  }

  const int b_ = bh >> 2;
  const int hh = bh & 3;
#pragma unroll
  for (int r = 0; r < 8; ++r) {
    int row = rb * 16 + r + 8 * hi;
    int node = b_ * N_NODES + row;
    float inv = 1.0f / l[r];
    out[(size_t)node * INNER + hh * DHEAD + ln]      = (_Float16)(o1[r] * inv);
    out[(size_t)node * INNER + hh * DHEAD + 16 + ln] = (_Float16)(o2[r] * inv);
  }
}

// ---------------------------------------------------------------- launch

extern "C" void kernel_launch(void* const* d_in, const int* in_sizes, int n_in,
                              void* d_out, int out_size, void* d_ws, size_t ws_size,
                              hipStream_t stream) {
  const float* x    = (const float*)d_in[0];
  const int*   ei   = (const int*)d_in[1];
  const float* ee   = (const float*)d_in[2];
  // d_in[3] = batch (unused; uniform graph sizes)
  const float* Wgcn = (const float*)d_in[4];
  const float* bgcn = (const float*)d_in[5];
  const float* root = (const float*)d_in[6];
  const float* Wq   = (const float*)d_in[7];
  const float* Wkv  = (const float*)d_in[8];
  const float* Wout = (const float*)d_in[9];
  const float* bout = (const float*)d_in[10];
  float* outp = (float*)d_out;

  char* ws = (char*)d_ws;
  // workspace layout (bytes), 256-aligned, with reuse:
  const size_t SZ_X16 = (size_t)TOTAL_NODES * DIM * 2;            // 8 MB
  _Float16* x16    = (_Float16*)(ws + 0);                         // reused as attn16
  _Float16* attn16 = x16;
  size_t off = SZ_X16;
  _Float16* wgcn_t = (_Float16*)(ws + off); off += 128 * 128 * 2;
  _Float16* wqkv_t = (_Float16*)(ws + off); off += 384 * 128 * 2;
  _Float16* wout_t = (_Float16*)(ws + off); off += 128 * 128 * 2;
  off = (off + 255) & ~(size_t)255;
  float* h = (float*)(ws + off);                                  // 16 MB, reused as q|k
  _Float16* qbuf = (_Float16*)(ws + off);
  _Float16* kbuf = (_Float16*)(ws + off + SZ_X16);
  off += (size_t)TOTAL_NODES * DIM * 4;
  float* degcnt = (float*)(ws + off); off += (size_t)TOTAL_NODES * 4;
  off = (off + 255) & ~(size_t)255;
  float* agg = (float*)(ws + off);                                // 16 MB, reused as vt
  _Float16* vtbuf = (_Float16*)(ws + off);
  off += (size_t)TOTAL_NODES * DIM * 4;
  _Float16* nf16 = (_Float16*)(ws + off); off += SZ_X16;

  const int ELTS = TOTAL_NODES * DIM;  // 4194304

  // 0) prep: x -> f16 ; weights -> transposed f16
  cvt_f32_to_f16_kernel<<<ELTS / 256, 256, 0, stream>>>(x, x16, ELTS);
  prep_wt_kernel<<<(128 * 128 + 255) / 256, 256, 0, stream>>>(Wgcn, wgcn_t, 128, 128);
  prep_wt_kernel<<<(128 * 128 + 255) / 256, 256, 0, stream>>>(Wq, wqkv_t, 128, 128);
  prep_wt_kernel<<<(128 * 256 + 255) / 256, 256, 0, stream>>>(Wkv, wqkv_t + 128 * 128, 128, 256);
  prep_wt_kernel<<<(128 * 128 + 255) / 256, 256, 0, stream>>>(Wout, wout_t, 128, 128);

  // 1) h = x @ W_gcn + b_gcn   (WMMA)
  gemm_f16_wmma_bias_kernel<<<(TOTAL_NODES / 16) * (DIM / 16) / 4, 128, 0, stream>>>(
      x16, wgcn_t, bgcn, h, TOTAL_NODES, DIM, DIM);

  // 2) zero atomic targets each call (graph-capture-safe)
  hipMemsetAsync(degcnt, 0, (size_t)TOTAL_NODES * 4, stream);
  hipMemsetAsync(agg, 0, (size_t)ELTS * 4, stream);

  // 3) degree counts + edge message scatter
  deg_count_kernel<<<TOTAL_EDGES / 256, 256, 0, stream>>>(ei, degcnt);
  edge_msg_kernel<<<TOTAL_EDGES * 32 / 256, 256, 0, stream>>>(ei, ee, h, degcnt, agg);

  // 4) combine -> nf (f16)
  combine_kernel<<<ELTS / 256, 256, 0, stream>>>(h, agg, degcnt, root, nf16);

  // 5) q,k,v = nf @ [Wq|Wkv]   (WMMA; v stored transposed)
  gemm_qkv_kernel<<<(TOTAL_NODES / 16) * 24 / 4, 128, 0, stream>>>(
      nf16, wqkv_t, qbuf, kbuf, vtbuf);

  // 6) flash attention per (b,h)   (WMMA x4 per key-chunk, async K/V staging)
  attn_kernel<<<B_GRAPHS * HEADS * 8, 128, 0, stream>>>(qbuf, kbuf, vtbuf, attn16);

  // 7) out = attn @ Wout + b_out   (WMMA, f32 out)
  gemm_f16_wmma_bias_kernel<<<(TOTAL_NODES / 16) * (DIM / 16) / 4, 128, 0, stream>>>(
      attn16, wout_t, bout, outp, TOTAL_NODES, DIM, DIM);
}
